// Phi3Attention_28544352650001
// MI455X (gfx1250) — matrix operand order repaired and checked by reading, not compiled, not run on hardware
//
#include <hip/hip_runtime.h>

// ---------------------------------------------------------------------------
// Types / helpers
// ---------------------------------------------------------------------------
typedef __attribute__((ext_vector_type(16))) __bf16 v16bf;
typedef __attribute__((ext_vector_type(8)))  float  v8f;

union Frag16 {               // 16 bf16 values = 8 dwords (A or B operand)
    v16bf        v;
    unsigned int u[8];
};

__device__ __forceinline__ unsigned short f2bf(float f) {
    union { float f; unsigned u; } x{f};
    unsigned r = x.u + 0x7FFFu + ((x.u >> 16) & 1u);   // round-to-nearest-even
    return (unsigned short)(r >> 16);
}

__device__ __forceinline__ v8f wmma_bf16(v16bf a, v16bf b, v8f c) {
    return __builtin_amdgcn_wmma_f32_16x16x32_bf16(
        /*neg_a=*/false, a, /*neg_b=*/false, b,
        /*c_mod=*/(short)0, c, /*reuse_a=*/false, /*reuse_b=*/false);
}

// CDNA5 async Global->LDS copy (16B per lane), tracked by ASYNCcnt.
// Inline asm: portable across ROCm-7.2 / amdgpu-toolchain builtin arity skew.
__device__ __forceinline__ void async_copy16(unsigned lds_byte,
                                             unsigned long long gaddr) {
    asm volatile("global_load_async_to_lds_b128 %0, %1, off"
                 :: "v"(lds_byte), "v"(gaddr) : "memory");
}
__device__ __forceinline__ void wait_async0() {
    asm volatile("s_wait_asynccnt 0" ::: "memory");
}

#define NEGBIG (-1000000000.0f)
#define SEQL   2048
#define DMODEL 3072
#define NHEAD  32
#define HDIM   96
#define QKVW   9216

// ---------------------------------------------------------------------------
// fp32 -> bf16 conversion (grid-stride)
// ---------------------------------------------------------------------------
__global__ void cvt_bf16_kernel(const float* __restrict__ src,
                                unsigned short* __restrict__ dst, size_t n) {
    size_t stride = (size_t)gridDim.x * blockDim.x;
    for (size_t i = (size_t)blockIdx.x * blockDim.x + threadIdx.x; i < n; i += stride)
        dst[i] = f2bf(src[i]);
}

// ---------------------------------------------------------------------------
// bf16 GEMM: C[M][N] (fp32) = A[M][K] * B[K][N], all row-major.
// Block tile 128x128, 8 waves (wave w -> rows [w*16, w*16+16)), K step 32.
// Double-buffered LDS; A tile staged via async Global->LDS (ASYNCcnt path),
// B tile staged transposed (manual, needs layout change for pair-gather).
// ---------------------------------------------------------------------------
__global__ __launch_bounds__(256) void gemm_bf16_kernel(
    const unsigned short* __restrict__ A,
    const unsigned short* __restrict__ B,
    float* __restrict__ C, int M, int N, int K) {

    __shared__ __align__(16) unsigned short As[2][128][32];  // [m][k], 16B rows*4
    __shared__ __align__(16) unsigned short Bs[2][128][34];  // [n][k] (transposed)

    const int n0   = blockIdx.x * 128;
    const int m0   = blockIdx.y * 128;
    const int tid  = threadIdx.x;
    const int lane = tid & 31;
    const int wave = tid >> 5;
    const int mrow = lane & 15;     // row within 16 (A) / col within 16 (B,C)
    const int lhalf = lane >> 4;    // half-wave selector

    auto stageA = [&](int buf, int k0) {
        // 128x32 ushort = 512 x 16B chunks, linear in both global row-chunks
        // and LDS (stride 32 elems = 64B rows).
        unsigned ldsbase = (unsigned)(size_t)(&As[buf][0][0]);
        for (int i = tid; i < 512; i += 256) {
            int r = i >> 2;            // 4 chunks per 64B row
            int c = (i & 3) * 8;       // element offset
            async_copy16(ldsbase + (unsigned)i * 16,
                         (unsigned long long)(size_t)(A + (size_t)(m0 + r) * K + k0 + c));
        }
    };
    auto stageB = [&](int buf, int k0) {
        for (int i = tid; i < 2048; i += 256) {
            int kk = i >> 6;
            int np = (i & 63) * 2;
            unsigned int w =
                *(const unsigned int*)(B + (size_t)(k0 + kk) * N + n0 + np);
            Bs[buf][np][kk]     = (unsigned short)(w & 0xFFFFu);
            Bs[buf][np + 1][kk] = (unsigned short)(w >> 16);
        }
    };

    v8f acc[8] = {};

    stageA(0, 0);
    stageB(0, 0);
    wait_async0();
    __syncthreads();

    const int nk = K / 32;
    for (int t = 0; t < nk; ++t) {
        const int cur = t & 1;
        if (t + 1 < nk) {              // prefetch next K-slab into other buffer
            stageA(cur ^ 1, (t + 1) * 32);
            stageB(cur ^ 1, (t + 1) * 32);
        }

        // --- A fragment for this wave (ISA 16-bit A 16x32 layout) ---
        Frag16 fa;
#pragma unroll
        for (int v = 0; v < 8; ++v) {
            int kf = ((v < 4) ? 0 : 16) + (v & 3) * 2 + lhalf * 8;
            fa.u[v] = *(const unsigned int*)&As[cur][wave * 16 + mrow][kf];
        }
        // --- 8 N-subtiles of 16 ---
#pragma unroll
        for (int j = 0; j < 8; ++j) {
            Frag16 fb;
#pragma unroll
            for (int v = 0; v < 8; ++v)
                fb.u[v] = *(const unsigned int*)&Bs[cur][j * 16 + mrow][((v < 4) ? 0 : 16) + (v & 3) * 2 + lhalf * 8];
            acc[j] = wmma_bf16(fa.v, fb.v, acc[j]);
        }

        wait_async0();                 // next slab's async copies done
        __syncthreads();               // + everyone finished reading cur
    }

    // --- store C (C/D layout: lane%16 = N, VGPR r = M row, +8 on upper half) ---
#pragma unroll
    for (int j = 0; j < 8; ++j)
#pragma unroll
        for (int r = 0; r < 8; ++r) {
            int row = m0 + wave * 16 + r + lhalf * 8;
            int col = n0 + j * 16 + mrow;
            C[(size_t)row * N + col] = acc[j][r];
        }
}

// ---------------------------------------------------------------------------
// RoPE + split: reads qkv fp32 [B][L][9216], writes
//   - roped k (fp32) and v (fp32) to d_out tuple slots  [B][32][L][96]
//   - bf16 q (scaled by HD^-1/2), k, v in head-major layout for attention
// ---------------------------------------------------------------------------
__global__ void rope_split_kernel(const float* __restrict__ qkv,
                                  const float* __restrict__ cosb,
                                  const float* __restrict__ sinb,
                                  float* __restrict__ kout,
                                  float* __restrict__ vout,
                                  unsigned short* __restrict__ qbf,
                                  unsigned short* __restrict__ kbf,
                                  unsigned short* __restrict__ vbf) {
    size_t t = (size_t)blockIdx.x * blockDim.x + threadIdx.x;
    const size_t total = (size_t)2 * SEQL * NHEAD * HDIM;
    if (t >= total) return;

    int i = (int)(t % HDIM);
    int h = (int)((t / HDIM) % NHEAD);
    int l = (int)((t / (HDIM * NHEAD)) % SEQL);
    int b = (int)(t / ((size_t)HDIM * NHEAD * SEQL));

    const float* row = qkv + (size_t)(b * SEQL + l) * QKVW;
    float c = cosb[l * HDIM + i];
    float s = sinb[l * HDIM + i];
    int   j = (i < HDIM / 2) ? i + HDIM / 2 : i - HDIM / 2;
    float sg = (i < HDIM / 2) ? -1.0f : 1.0f;

    size_t oidx = ((size_t)((b * NHEAD + h) * SEQL + l)) * HDIM + i;

    // q (fold in softmax scale HD^-0.5)
    float qv = row[h * HDIM + i];
    float qp = row[h * HDIM + j];
    float qr = qv * c + sg * qp * s;
    qbf[oidx] = f2bf(qr * 0.10206207261596575f);

    // k (roped; also a tuple output)
    float kv = row[DMODEL + h * HDIM + i];
    float kp = row[DMODEL + h * HDIM + j];
    float kr = kv * c + sg * kp * s;
    kout[oidx] = kr;
    kbf[oidx]  = f2bf(kr);

    // v (tuple output)
    float vv = row[2 * DMODEL + h * HDIM + i];
    vout[oidx] = vv;
    vbf[oidx]  = f2bf(vv);
}

// ---------------------------------------------------------------------------
// Causal flash attention, one block per (b*h, 128 q-rows), 8 waves x 16 rows.
// bf16 WMMA for Q*K^T and P*V with fp32 online-softmax accumulation.
// K tile staged via async Global->LDS (contiguous 6KB), V staged transposed.
// ---------------------------------------------------------------------------
__global__ __launch_bounds__(256) void flash_attn_kernel(
    const unsigned short* __restrict__ qbf,
    const unsigned short* __restrict__ kbf,
    const unsigned short* __restrict__ vbf,
    unsigned short* __restrict__ obf) {

    __shared__ __align__(16) unsigned short Ks[32][96];      // [key][hd]
    __shared__ __align__(16) unsigned short Vt[96][34];      // [hd][key]
    __shared__ __align__(16) unsigned short Ps[8][16][34];   // per-wave P tile

    const int bh   = blockIdx.y;          // b*32 + h
    const int q0   = blockIdx.x * 128;
    const int tid  = threadIdx.x;
    const int lane = tid & 31;
    const int wave = tid >> 5;
    const int mrow = lane & 15;
    const int lhalf = lane >> 4;

    const unsigned short* qhead = qbf + (size_t)bh * SEQL * HDIM;
    const unsigned short* khead = kbf + (size_t)bh * SEQL * HDIM;
    const unsigned short* vhead = vbf + (size_t)bh * SEQL * HDIM;

    const int qbase = q0 + wave * 16;

    // Q fragments (3 chunks of K=32 covering HD=96), gathered straight from global
    Frag16 qa[3];
#pragma unroll
    for (int c3 = 0; c3 < 3; ++c3)
#pragma unroll
        for (int v = 0; v < 8; ++v) {
            int kf = ((v < 4) ? 0 : 16) + (v & 3) * 2 + lhalf * 8;
            qa[c3].u[v] = *(const unsigned int*)(qhead +
                          (size_t)(qbase + mrow) * HDIM + c3 * 32 + kf);
        }

    v8f   o[6] = {};
    float mrun[8], lrun[8];
#pragma unroll
    for (int r = 0; r < 8; ++r) { mrun[r] = -1e30f; lrun[r] = 0.0f; }

    const unsigned ksbase = (unsigned)(size_t)(&Ks[0][0]);

    const int nChunks = blockIdx.x * 4 + 4;   // causal: keys <= q0+127
    for (int kc = 0; kc < nChunks; ++kc) {
        const int kstart = kc * 32;
        __syncthreads();
        // stage K tile: contiguous 6KB -> async Global->LDS, 16B per lane
        {
            unsigned long long ksrc =
                (unsigned long long)(size_t)(khead + (size_t)kstart * HDIM);
            for (int i = tid; i < 384; i += 256)
                async_copy16(ksbase + (unsigned)i * 16, ksrc + (unsigned)i * 16);
        }
        // stage V tile transposed (manual: async path cannot transpose)
        for (int i = tid; i < 1536; i += 256) {
            int r = i / 48;
            int c = (i % 48) * 2;
            unsigned int w =
                *(const unsigned int*)(vhead + (size_t)(kstart + r) * HDIM + c);
            Vt[c][r]     = (unsigned short)(w & 0xFFFFu);
            Vt[c + 1][r] = (unsigned short)(w >> 16);
        }
        wait_async0();
        __syncthreads();

        // scores: two 16-key fragments, HD=96 -> 3 WMMAs each
        v8f s[2];
#pragma unroll
        for (int j = 0; j < 2; ++j) {
            v8f cf = {};
#pragma unroll
            for (int c3 = 0; c3 < 3; ++c3) {
                Frag16 fb;
#pragma unroll
                for (int v = 0; v < 8; ++v)
                    fb.u[v] = *(const unsigned int*)
                        &Ks[j * 16 + mrow][c3 * 32 + ((v < 4) ? 0 : 16) + (v & 3) * 2 + lhalf * 8];
                cf = wmma_bf16(qa[c3].v, fb.v, cf);
            }
            s[j] = cf;
        }

        // causal mask (lane%16 = key col, VGPR r (+8 upper half) = q row)
#pragma unroll
        for (int j = 0; j < 2; ++j) {
            int kg = kstart + j * 16 + mrow;
#pragma unroll
            for (int r = 0; r < 8; ++r) {
                int qr = qbase + r + lhalf * 8;
                if (kg > qr) s[j][r] = NEGBIG;
            }
        }

        // online softmax
        float pv[2][8];
#pragma unroll
        for (int r = 0; r < 8; ++r) {
            float mx = fmaxf(s[0][r], s[1][r]);
            mx = fmaxf(mx, __shfl_xor(mx, 1));
            mx = fmaxf(mx, __shfl_xor(mx, 2));
            mx = fmaxf(mx, __shfl_xor(mx, 4));
            mx = fmaxf(mx, __shfl_xor(mx, 8));
            float mnew  = fmaxf(mrun[r], mx);
            float alpha = __expf(mrun[r] - mnew);
            float p0 = __expf(s[0][r] - mnew);
            float p1 = __expf(s[1][r] - mnew);
            float ps = p0 + p1;
            ps += __shfl_xor(ps, 1);
            ps += __shfl_xor(ps, 2);
            ps += __shfl_xor(ps, 4);
            ps += __shfl_xor(ps, 8);
            lrun[r] = lrun[r] * alpha + ps;
            mrun[r] = mnew;
#pragma unroll
            for (int f = 0; f < 6; ++f) o[f][r] *= alpha;
            pv[0][r] = p0;
            pv[1][r] = p1;
        }

        // P -> LDS (bf16) so it can be re-gathered as an A fragment
#pragma unroll
        for (int j = 0; j < 2; ++j)
#pragma unroll
            for (int r = 0; r < 8; ++r)
                Ps[wave][r + lhalf * 8][j * 16 + mrow] = f2bf(pv[j][r]);
        __syncthreads();

        Frag16 fp;
#pragma unroll
        for (int v = 0; v < 8; ++v) {
            int kf = ((v < 4) ? 0 : 16) + (v & 3) * 2 + lhalf * 8;
            fp.u[v] = *(const unsigned int*)&Ps[wave][mrow][kf];
        }
        // P (16x32 keys) * V (32 keys x 96 hd) -> 6 output fragments
#pragma unroll
        for (int f = 0; f < 6; ++f) {
            Frag16 fb;
#pragma unroll
            for (int v = 0; v < 8; ++v)
                fb.u[v] = *(const unsigned int*)&Vt[f * 16 + mrow][((v < 4) ? 0 : 16) + (v & 3) * 2 + lhalf * 8];
            o[f] = wmma_bf16(fp.v, fb.v, o[f]);
        }
    }

    // epilogue: normalize and store bf16 attention output [B][L][H*HD]
    const int b = bh >> 5;
    const int h = bh & 31;
#pragma unroll
    for (int f = 0; f < 6; ++f)
#pragma unroll
        for (int r = 0; r < 8; ++r) {
            int row = qbase + r + lhalf * 8;
            float val = o[f][r] / lrun[r];
            obf[(size_t)(b * SEQL + row) * DMODEL + h * HDIM + f * 16 + mrow] =
                f2bf(val);
        }
}

// ---------------------------------------------------------------------------
// Launch
// ---------------------------------------------------------------------------
extern "C" void kernel_launch(void* const* d_in, const int* in_sizes, int n_in,
                              void* d_out, int out_size, void* d_ws, size_t ws_size,
                              hipStream_t stream) {
    (void)in_sizes; (void)n_in; (void)out_size; (void)ws_size;

    const float* x     = (const float*)d_in[0];
    const float* cosb  = (const float*)d_in[1];
    const float* sinb  = (const float*)d_in[2];
    /* d_in[3] = mask: causal mask computed analytically in-kernel */
    const float* w_qkv = (const float*)d_in[4];
    const float* w_o   = (const float*)d_in[5];

    float* outp = (float*)d_out;                                  // (B,L,3072)
    float* kout = outp + (size_t)2 * SEQL * DMODEL;               // (B,32,L,96)
    float* vout = kout + (size_t)2 * NHEAD * SEQL * HDIM;         // (B,32,L,96)

    char* ws = (char*)d_ws;
    unsigned short* xbf    = (unsigned short*)(ws + 0);           // also reused as attn out bf16
    unsigned short* wqkvbf = (unsigned short*)(ws + 25165824);
    unsigned short* wobf   = (unsigned short*)(ws + 81788928);
    float*          qkv    = (float*)        (ws + 100663296);
    unsigned short* qbf    = (unsigned short*)(ws + 251658240);
    unsigned short* kbf    = (unsigned short*)(ws + 276824064);
    unsigned short* vbf    = (unsigned short*)(ws + 301989888);

    // 1) fp32 -> bf16 operands
    cvt_bf16_kernel<<<4096, 256, 0, stream>>>(x,     xbf,    (size_t)4096 * DMODEL);
    cvt_bf16_kernel<<<8192, 256, 0, stream>>>(w_qkv, wqkvbf, (size_t)DMODEL * QKVW);
    cvt_bf16_kernel<<<4096, 256, 0, stream>>>(w_o,   wobf,   (size_t)DMODEL * DMODEL);

    // 2) qkv = x @ w_qkv   (4096 x 9216 x 3072)
    dim3 g1(QKVW / 128, 4096 / 128);
    gemm_bf16_kernel<<<g1, 256, 0, stream>>>(xbf, wqkvbf, qkv, 4096, QKVW, DMODEL);

    // 3) RoPE + split (also emits tuple outputs k, v)
    rope_split_kernel<<<(2 * SEQL * NHEAD * HDIM + 255) / 256, 256, 0, stream>>>(
        qkv, cosb, sinb, kout, vout, qbf, kbf, vbf);

    // 4) causal flash attention -> bf16 context (reuses xbf region)
    dim3 g2(SEQL / 128, 2 * NHEAD);
    flash_attn_kernel<<<g2, 256, 0, stream>>>(qbf, kbf, vbf, xbf);

    // 5) out = context @ w_o  (4096 x 3072 x 3072)
    dim3 g3(DMODEL / 128, 4096 / 128);
    gemm_bf16_kernel<<<g3, 256, 0, stream>>>(xbf, wobf, outp, 4096, DMODEL, DMODEL);
}
